// AANet_influent_51874615001474
// MI455X (gfx1250) — matrix operand order repaired and checked by reading, not compiled, not run on hardware
//
#include <hip/hip_runtime.h>
#include <hip/hip_bf16.h>

typedef __attribute__((ext_vector_type(16))) __bf16 v16bf;
typedef __attribute__((ext_vector_type(8)))  __bf16 v8bf;
typedef __attribute__((ext_vector_type(8)))  float  v8f;

// ---------------------------------------------------------------------------
// Fragment loaders (CDNA5 WMMA bf16 16x16x32 operand layouts, wave32)
// ---------------------------------------------------------------------------
__device__ __forceinline__ v16bf load_afrag(const __bf16* __restrict__ A,
                                            int row, int K, int k0, int hi)
{
    // A (16x32, MxK): lane holds M=lane%16; lane half selects K
    // 0..15 (hi=0) vs 16..31 (hi=1): two contiguous 8-half loads.
    const __bf16* p = A + (size_t)row * K + k0 + hi * 16;
    v8bf lo = *(const v8bf*)p;
    v8bf hh = *(const v8bf*)(p + 8);
    v16bf r;
#pragma unroll
    for (int i = 0; i < 8; ++i) { r[i] = lo[i]; r[i + 8] = hh[i]; }
    return r;
}

__device__ __forceinline__ v16bf load_bfrag(const __bf16* __restrict__ W,
                                            int ncol, int K, int k0, int hi)
{
    // B (32x16, KxN): lane holds N=lane%16; lane half selects K 0..15 vs
    // 16..31 -> 16 contiguous bf16 (32B) from the packed [Np,K] weight row.
    return *(const v16bf*)(W + (size_t)ncol * K + k0 + hi * 16);
}

// ---------------------------------------------------------------------------
// WMMA GEMM:  Y[M,N] = A[M,K] * W^T  (+bias, optional ReLU)
//   A   : bf16 [M, K], K % 32 == 0
//   W   : bf16 [Np, K] packed (Np = ceil32(N)), zero padded
//   Yf  : fp32 [M, N] (optional)   Ybf : bf16 [M, N] (optional)
// Wave tile 64x32 (4 M-subtiles x 2 N-subtiles; A reused x2, B reused x4).
// K loop manually unrolled x2 with explicit ping/pong fragment sets: no
// register rotation copies; loads of step k+1 overlap WMMAs of step k.
// Block = 8 waves -> 512x32.  grid = (M/512, Np/32).
// ---------------------------------------------------------------------------
__global__ __launch_bounds__(256) void gemm_wmma(
    const __bf16* __restrict__ A, const __bf16* __restrict__ W,
    const float* __restrict__ bias, float* __restrict__ Yf,
    __bf16* __restrict__ Ybf, int M, int K, int N, int relu)
{
    const int lane  = threadIdx.x & 31;
    const int wave  = threadIdx.x >> 5;
    const int mBase = blockIdx.x * 512 + wave * 64;
    const int nBase = blockIdx.y * 32;
    const int col16 = lane & 15;
    const int hi    = lane >> 4;

    v8f acc[4][2] = {};
    v16bf a0[4], b0[2], a1[4], b1[2];

    auto loadStep = [&](v16bf (&a)[4], v16bf (&b)[2], int k) {
#pragma unroll
        for (int s = 0; s < 4; ++s)
            a[s] = load_afrag(A, mBase + 16 * s + col16, K, k, hi);
#pragma unroll
        for (int j = 0; j < 2; ++j)
            b[j] = load_bfrag(W, nBase + 16 * j + col16, K, k, hi);
    };
    auto mmaStep = [&](const v16bf (&a)[4], const v16bf (&b)[2]) {
#pragma unroll
        for (int s = 0; s < 4; ++s)
#pragma unroll
            for (int j = 0; j < 2; ++j)
                acc[s][j] = __builtin_amdgcn_wmma_f32_16x16x32_bf16(
                    false, a[s], false, b[j], (short)0, acc[s][j], false, false);
    };

    loadStep(a0, b0, 0);
    int k0 = 0;
    while (true) {
        int kn = k0 + 32;                 // phase 0: consume set0, prefetch set1
        if (kn < K) loadStep(a1, b1, kn);
        mmaStep(a0, b0);
        k0 = kn;
        if (k0 >= K) break;
        kn = k0 + 32;                     // phase 1: consume set1, prefetch set0
        if (kn < K) loadStep(a0, b0, kn);
        mmaStep(a1, b1);
        k0 = kn;
        if (k0 >= K) break;
    }

#pragma unroll
    for (int j = 0; j < 2; ++j) {
        const int col = nBase + 16 * j + col16;
        if (col < N) {
            const float bv = bias ? bias[col] : 0.0f;
#pragma unroll
            for (int s = 0; s < 4; ++s) {
#pragma unroll
                for (int v = 0; v < 8; ++v) {
                    int row = mBase + 16 * s + v + 8 * hi;   // C/D layout
                    float val = acc[s][j][v] + bv;
                    if (relu) val = fmaxf(val, 0.0f);
                    if (Yf)  Yf[(size_t)row * N + col] = val;
                    if (Ybf) Ybf[(size_t)row * N + col] = (__bf16)val;
                }
            }
        }
    }
}

// Pack conv/linear weight: take middle tap of [cout, cin, k] fp32, emit
// bf16 [Np, Kp] zero padded (Np=ceil32, Kp=ceil32).
__global__ void pack_w(const float* __restrict__ W, __bf16* __restrict__ out,
                       int cout, int cin, int k, int Np, int Kp)
{
    int i = blockIdx.x * blockDim.x + threadIdx.x;
    if (i >= Np * Kp) return;
    int n = i / Kp, c = i % Kp;
    float v = (n < cout && c < cin)
                  ? W[((size_t)n * cin + c) * k + (k - 1) / 2] : 0.0f;
    out[i] = (__bf16)v;
}

// Haar db1 level-1 approx along feature axis: [B,128] fp32 -> [B,64] bf16
__global__ void haar_k(const float* __restrict__ ne, __bf16* __restrict__ out,
                       int total)
{
    int i = blockIdx.x * blockDim.x + threadIdx.x;
    if (i >= total) return;
    int b = i >> 6, j = i & 63;
    float v = (ne[(size_t)b * 128 + 2 * j] + ne[(size_t)b * 128 + 2 * j + 1])
              * 0.70710678118654752440f;
    out[i] = (__bf16)v;
}

// fp32 [B,N] -> bf16 [B,Np] with zero pad (for K=130 -> 160)
__global__ void cvt_pad(const float* __restrict__ in, __bf16* __restrict__ out,
                        int Bn, int N, int Np)
{
    int i = blockIdx.x * blockDim.x + threadIdx.x;
    if (i >= Bn * Np) return;
    int b = i / Np, c = i % Np;
    out[i] = (__bf16)((c < N) ? in[(size_t)b * N + c] : 0.0f);
}

// BN stats stage 1: coalesced partial sums.
// grid = (N/64, chunks); block = 256 = 64 channels x 4 row-groups.
__global__ void bn_stats_part(const float* __restrict__ Y,
                              float* __restrict__ part,   // [chunks][N][2]
                              int N, int rowsPerChunk)
{
    const int c     = blockIdx.x * 64 + (threadIdx.x & 63);
    const int rg    = threadIdx.x >> 6;      // 0..3
    const int chunk = blockIdx.y;
    const int b0    = chunk * rowsPerChunk;
    float s = 0.0f, sq = 0.0f;
    for (int b = b0 + rg; b < b0 + rowsPerChunk; b += 4) {
        float v = Y[(size_t)b * N + c];      // consecutive c across lanes
        s += v; sq += v * v;
    }
    __shared__ float sh0[256], sh1[256];
    sh0[threadIdx.x] = s; sh1[threadIdx.x] = sq;
    __syncthreads();
    if (rg == 0) {
#pragma unroll
        for (int r = 1; r < 4; ++r) {
            s  += sh0[threadIdx.x + 64 * r];
            sq += sh1[threadIdx.x + 64 * r];
        }
        size_t o = ((size_t)chunk * N + c) * 2;
        part[o] = s; part[o + 1] = sq;
    }
}

// BN stats stage 2: reduce chunks per channel -> mean, rstd.
__global__ void bn_stats_final(const float* __restrict__ part,
                               float* __restrict__ stats,
                               int N, int chunks, int Bn)
{
    int c = blockIdx.x * blockDim.x + threadIdx.x;
    if (c >= N) return;
    float s = 0.0f, sq = 0.0f;
    for (int ch = 0; ch < chunks; ++ch) {
        size_t o = ((size_t)ch * N + c) * 2;
        s += part[o]; sq += part[o + 1];
    }
    float m   = s / (float)Bn;
    float var = sq / (float)Bn - m * m;
    stats[2 * c]     = m;
    stats[2 * c + 1] = rsqrtf(var + 1e-5f);
}

// out = alpha * relu(g*(y-m)*rstd + beta) + (add ? add : 0)
__global__ void bn_apply(const float* __restrict__ Y, const float* __restrict__ st,
                         const float* __restrict__ g, const float* __restrict__ be,
                         const float* __restrict__ add, float alpha,
                         __bf16* __restrict__ obf, float* __restrict__ of,
                         int total, int N)
{
    int i = blockIdx.x * blockDim.x + threadIdx.x;
    if (i >= total) return;
    int c = i % N;
    float v = g[c] * (Y[i] - st[2 * c]) * st[2 * c + 1] + be[c];
    v = fmaxf(v, 0.0f) * alpha;
    if (add) v += add[i];
    if (obf) obf[i] = (__bf16)v;
    if (of)  of[i]  = v;
}

// Row softmax for tiny N (<=8)
__global__ void softmax_rows(const float* __restrict__ Y, float* __restrict__ out,
                             int Bn, int N)
{
    int b = blockIdx.x * blockDim.x + threadIdx.x;
    if (b >= Bn) return;
    float mx = -3.4e38f;
    for (int j = 0; j < N; ++j) mx = fmaxf(mx, Y[(size_t)b * N + j]);
    float e[8], s = 0.0f;
    for (int j = 0; j < N; ++j) { e[j] = expf(Y[(size_t)b * N + j] - mx); s += e[j]; }
    float inv = 1.0f / s;
    for (int j = 0; j < N; ++j) out[(size_t)b * N + j] = e[j] * inv;
}

// feature = gf*wo[:,0] + tf*wo[:,1]  (fp32 to d_out slice + bf16 for MLP heads)
__global__ void fuse_feat(const float* __restrict__ gf, const float* __restrict__ tf,
                          const float* __restrict__ wo, float* __restrict__ fout,
                          __bf16* __restrict__ fbf, int total)
{
    int i = blockIdx.x * blockDim.x + threadIdx.x;
    if (i >= total) return;
    int b = i >> 9;   // /512
    float v = gf[i] * wo[2 * b] + tf[i] * wo[2 * b + 1];
    fout[i] = v;
    fbf[i]  = (__bf16)v;
}

// ---------------------------------------------------------------------------
extern "C" void kernel_launch(void* const* d_in, const int* in_sizes, int n_in,
                              void* d_out, int out_size, void* d_ws, size_t ws_size,
                              hipStream_t stream)
{
    (void)in_sizes; (void)n_in; (void)out_size; (void)ws_size;
    const int Bn = 16384;
    const int CHUNKS = 64, ROWS_PER_CHUNK = Bn / CHUNKS;
    auto inp = [&](int i) { return (const float*)d_in[i]; };

    // ---- bump allocator in d_ws (same sequence every call => deterministic)
    char* ws = (char*)d_ws;
    size_t off = 0;
    auto alloc = [&](size_t bytes) -> void* {
        off = (off + 255) & ~(size_t)255;
        void* p = ws + off;
        off += bytes;
        return p;
    };

    float*  Ybuf   = (float*) alloc((size_t)Bn * 1024 * sizeof(float));
    __bf16* actA   = (__bf16*)alloc((size_t)Bn * 1024 * sizeof(__bf16));
    __bf16* actB   = (__bf16*)alloc((size_t)Bn * 1024 * sizeof(__bf16));
    __bf16* xbf    = (__bf16*)alloc((size_t)Bn * 128  * sizeof(__bf16));
    float*  gf32   = (float*) alloc((size_t)Bn * 512  * sizeof(float));
    float*  tf32   = (float*) alloc((size_t)Bn * 512  * sizeof(float));
    __bf16* featbf = (__bf16*)alloc((size_t)Bn * 512  * sizeof(__bf16));
    float*  wo     = (float*) alloc((size_t)Bn * 2    * sizeof(float));
    float*  statsB = (float*) alloc(1024 * 2 * sizeof(float));
    float*  partB  = (float*) alloc((size_t)CHUNKS * 1024 * 2 * sizeof(float));

    auto packW = [&](int widx, int cout, int cin, int k) -> const __bf16* {
        int Np = (cout + 31) & ~31, Kp = (cin + 31) & ~31;
        __bf16* p = (__bf16*)alloc((size_t)Np * Kp * sizeof(__bf16));
        int tot = Np * Kp;
        pack_w<<<(tot + 255) / 256, 256, 0, stream>>>(inp(widx), p, cout, cin, k, Np, Kp);
        return p;
    };

    auto gemm = [&](const __bf16* A, const __bf16* Wp, const float* bias,
                    float* Yf, __bf16* Ybf, int Kp, int N, int relu) {
        int Np = (N + 31) & ~31;
        dim3 grid(Bn / 512, Np / 32);
        gemm_wmma<<<grid, 256, 0, stream>>>(A, Wp, bias, Yf, Ybf, Bn, Kp, N, relu);
    };

    // Conv(k) + BatchNorm(train) + ReLU  [+ alpha / add fusions]
    auto cbr = [&](const __bf16* A, int pb, int cin, int cout, int k,
                   const float* add, float alpha, __bf16* obf, float* of) {
        const __bf16* Wp = packW(pb, cout, cin, k);
        int Kp = (cin + 31) & ~31;
        gemm(A, Wp, inp(pb + 1), Ybuf, nullptr, Kp, cout, 0);
        dim3 g1(cout / 64, CHUNKS);
        bn_stats_part<<<g1, 256, 0, stream>>>(Ybuf, partB, cout, ROWS_PER_CHUNK);
        bn_stats_final<<<(cout + 255) / 256, 256, 0, stream>>>(partB, statsB, cout, CHUNKS, Bn);
        int tot = Bn * cout;
        bn_apply<<<(tot + 255) / 256, 256, 0, stream>>>(
            Ybuf, statsB, inp(pb + 2), inp(pb + 3), add, alpha, obf, of, tot, cout);
    };

    // Input index map (setup_inputs flattening order):
    // 0 node_embedding, 1 weight_input, gen 2..33 (8x4), tgt 34..41 (2x4),
    // wave 42..45, weight 46..53 (2x4), res1 54..57, res2 58..61,
    // sample_mlp 62..71 (5x2), link_mlp 72..81, net_mlp 82..89 (4x2),
    // wsm 90..91, infl 92..95, coeff 96 (unused: grad_reverse fwd = identity)

    // 1) Haar approx -> [B,64] bf16
    haar_k<<<(Bn * 64 + 255) / 256, 256, 0, stream>>>(inp(0), actA, Bn * 64);
    // 2) wave CBR(64->128); x = node_embedding + 0.3*wave  -> xbf
    cbr(actA, 42, 64, 128, 3, inp(0), 0.3f, xbf, nullptr);

    // 3) gen stack (8 CBR layers), last also keeps fp32 copy for residual
    static const int genspec[8][2] = {{128,256},{256,512},{512,1024},{1024,512},
                                      {512,512},{512,1024},{1024,512},{512,512}};
    const __bf16* cur = xbf;
    for (int l = 0; l < 8; ++l) {
        __bf16* nxt = (l & 1) ? actA : actB;
        cbr(cur, 2 + 4 * l, genspec[l][0], genspec[l][1], 3,
            nullptr, 1.0f, nxt, (l == 7) ? gf32 : nullptr);
        cur = nxt;
    }
    // res1: gf = CBR_k1(gf) + gf   (in-place fp32)
    cbr(cur, 54, 512, 512, 1, gf32, 1.0f, nullptr, gf32);

    // 4) tgt stack + res2
    cbr(xbf,  34, 128, 256, 3, nullptr, 1.0f, actA, nullptr);
    cbr(actA, 38, 256, 512, 3, nullptr, 1.0f, actB, tf32);
    cbr(actB, 58, 512, 512, 1, tf32, 1.0f, nullptr, tf32);

    // 5) weight path: [B,130] -> pad 160 -> 2 CBR -> wsm linear -> softmax2
    cvt_pad<<<(Bn * 160 + 255) / 256, 256, 0, stream>>>(inp(1), actA, Bn, 130, 160);
    cbr(actA, 46, 130, 256, 3, nullptr, 1.0f, actB, nullptr);
    cbr(actB, 50, 256, 512, 3, nullptr, 1.0f, actA, nullptr);
    {
        const __bf16* Wp = packW(90, 2, 512, 1);
        gemm(actA, Wp, inp(91), Ybuf, nullptr, 512, 2, 0);
        softmax_rows<<<(Bn + 255) / 256, 256, 0, stream>>>(Ybuf, wo, Bn, 2);
    }

    // 6) feature = gf*wo0 + tf*wo1 -> d_out slice (fp32) + bf16 for heads
    float* out_f = (float*)d_out;
    float* featO = out_f + (size_t)Bn * 9;      // after link(2)+net(5)+sample(2)
    fuse_feat<<<(Bn * 512 + 255) / 256, 256, 0, stream>>>(
        gf32, tf32, wo, featO, featbf, Bn * 512);

    // 7) MLP heads (relu fused in GEMM between layers; last layer -> fp32)
    auto mlp = [&](int base, const int* dims, int nl, float* finalY) {
        const __bf16* c = featbf;
        for (int i = 0; i < nl; ++i) {
            const __bf16* Wp = packW(base + 2 * i, dims[i + 1], dims[i], 1);
            if (i < nl - 1) {
                __bf16* nxt = (i & 1) ? actB : actA;
                gemm(c, Wp, inp(base + 2 * i + 1), nullptr, nxt, dims[i], dims[i + 1], 1);
                c = nxt;
            } else {
                gemm(c, Wp, inp(base + 2 * i + 1), finalY, nullptr, dims[i], dims[i + 1], 0);
            }
        }
    };
    static const int ldims[6] = {512, 256, 128, 64, 32, 2};
    static const int ndims[5] = {512, 256, 128, 64, 5};
    static const int sdims[6] = {512, 256, 128, 64, 32, 2};
    static const int idims[3] = {512, 128, 4};

    mlp(72, ldims, 5, Ybuf);   // link
    softmax_rows<<<(Bn + 255) / 256, 256, 0, stream>>>(Ybuf, out_f, Bn, 2);
    mlp(82, ndims, 4, Ybuf);   // network
    softmax_rows<<<(Bn + 255) / 256, 256, 0, stream>>>(Ybuf, out_f + (size_t)Bn * 2, Bn, 5);
    mlp(62, sdims, 5, Ybuf);   // sample
    softmax_rows<<<(Bn + 255) / 256, 256, 0, stream>>>(Ybuf, out_f + (size_t)Bn * 7, Bn, 2);
    mlp(92, idims, 2, out_f + (size_t)Bn * 521);   // influence (no softmax)
}
